// ContrastMemLoss_16509854286610
// MI455X (gfx1250) — matrix operand order repaired and checked by reading, not compiled, not run on hardware
//
#include <hip/hip_runtime.h>
#include <hip/hip_bf16.h>

// ---------------------------------------------------------------------------
// ContrastMemLoss for MI455X (gfx1250, wave32, WMMA).
//
// Pipeline:
//  1) sum_kernel      : S1[c] = sum_p x[c,p]                    (per net)
//  2) cov_kernel      : G = X X^T via bf16 WMMA, split-K atomics (per net)
//  3) stats_kernel    : mu_d = w_d . mean(x);  rsig_d = rsqrt(w_d Cov w_d^T + eps)
//                       (b1 cancels inside BN, so it is never needed)
//  4) embed_kernel    : for 3952 sampled pixels: W1 GEMM -> BN -> ReLU ->
//                       W2 GEMM -> L2 norm -> bf16 features (WMMA, per net)
//  5) logits_kernel   : L = FS FT^T / TEMP (bf16 WMMA), stores L and L^T
//  6) row_loss_kernel : per-row max / neg-sum / positive log-prob (x2 dirs)
//  7) final_kernel    : out = 0.5 * (lossA + lossB)
// ---------------------------------------------------------------------------

typedef __bf16 bf16_t;
typedef __attribute__((ext_vector_type(16))) __bf16 v16bf;
typedef __attribute__((ext_vector_type(8)))  __bf16 v8bf;
typedef __attribute__((ext_vector_type(8)))  float   v8f;
typedef __attribute__((ext_vector_type(4)))  float   f32x4;

__device__ inline bf16_t f2bf(float f) { return (bf16_t)f; }

__device__ inline v16bf cat8(v8bf lo, v8bf hi) {
  return __builtin_shufflevector(lo, hi, 0,1,2,3,4,5,6,7,8,9,10,11,12,13,14,15);
}

// A fragment (16x32 bf16, MxK): row-major buffer [row][k], ldk elements.
// Per ISA layout: lanes 0-15 -> K {0..7,16..23}, lanes 16-31 -> K {8..15,24..31}.
__device__ inline v16bf frag_a_lds(const bf16_t* tile, int ldk, int kc, int lane) {
  int off = ((lane >> 4) & 1) * 8;
  const bf16_t* p = tile + (size_t)(lane & 15) * ldk + kc + off;
  v8bf lo = *(const v8bf*)p;
  v8bf hi = *(const v8bf*)(p + 16);
  return cat8(lo, hi);
}

// B fragment (32x16 bf16, KxN): B[k][n] stored as [n][k] row-major.
// lanes 0-15: K {0..7,16..23} at column n=lane; lanes 16-31: K {8..15,24..31}, n=lane-16.
__device__ inline v16bf frag_b_lds(const bf16_t* tile, int ldk, int kc, int lane) {
  int koff = ((lane >> 4) & 1) * 8;
  const bf16_t* p = tile + (size_t)(lane & 15) * ldk + kc + koff;
  v8bf lo = *(const v8bf*)p;
  v8bf hi = *(const v8bf*)(p + 16);
  return cat8(lo, hi);
}

// A fragment built from a global f32 row (on-the-fly bf16 conversion).
__device__ inline v16bf frag_a_gf32(const float* rowbase, int kc, int lane) {
  int off = ((lane >> 4) & 1) * 8;
  const float* p = rowbase + kc + off;
  f32x4 c0 = *(const f32x4*)(p);
  f32x4 c1 = *(const f32x4*)(p + 4);
  f32x4 c2 = *(const f32x4*)(p + 16);
  f32x4 c3 = *(const f32x4*)(p + 20);
  v16bf r = {};
#pragma unroll
  for (int k = 0; k < 4; ++k) {
    r[k]      = f2bf(c0[k]);
    r[4 + k]  = f2bf(c1[k]);
    r[8 + k]  = f2bf(c2[k]);
    r[12 + k] = f2bf(c3[k]);
  }
  return r;
}

__device__ inline v8f wmma_bf16(v16bf a, v16bf b, v8f c) {
  return __builtin_amdgcn_wmma_f32_16x16x32_bf16(false, a, false, b, (short)0, c,
                                                 false, false);
}

__device__ inline float block_reduce_sum(float v, float* red) {
  int t = threadIdx.x;
  red[t] = v; __syncthreads();
  for (int s = 128; s > 0; s >>= 1) { if (t < s) red[t] += red[t + s]; __syncthreads(); }
  float r = red[0]; __syncthreads();
  return r;
}

// ---------------------------------------------------------------------------
__global__ void zero_kernel(float* p, size_t n) {
  size_t i = (size_t)blockIdx.x * blockDim.x + threadIdx.x;
  size_t stride = (size_t)gridDim.x * blockDim.x;
  for (; i < n; i += stride) p[i] = 0.0f;
}

// S1[c] = sum over all pixels of x[n,c,:,:]
__global__ void sum_kernel(const float* pS, const float* pT, float* S1) {
  int row = blockIdx.x;               // n*512 + c, 0..4095
  int net = blockIdx.y;
  const float* X = net ? pT : pS;
  const float* p = X + (size_t)row * 16384;
  float s = 0.f;
  for (int j = threadIdx.x * 4; j < 16384; j += 1024) {
    f32x4 v = *(const f32x4*)(p + j);
    s += v[0] + v[1] + v[2] + v[3];
  }
  __shared__ float red[256];
  s = block_reduce_sum(s, red);
  if (threadIdx.x == 0) atomicAdd(&S1[net * 512 + (row & 511)], s);
}

// G = X X^T (512x512), bf16 WMMA, split-K over 64 slices (2048 p each).
__global__ void cov_kernel(const float* pS, const float* pT, float* Gws) {
  __shared__ bf16_t As[64][64];
  __shared__ bf16_t Bs[64][64];
  int net = blockIdx.z;
  const float* X = net ? pT : pS;
  float* G = Gws + (size_t)net * 512 * 512;
  int ib = blockIdx.x >> 3, jb = blockIdx.x & 7;
  int slice = blockIdx.y;
  int n = slice >> 3;
  int hw0 = (slice & 7) * 2048;
  int t = threadIdx.x, lane = t & 31, wave = t >> 5;
  int rr = t >> 2, c4 = (t & 3) * 16;
  const size_t HW = 16384;
  int t0 = wave * 2, t1 = t0 + 1;
  int ti0 = t0 >> 2, tj0 = t0 & 3, ti1 = t1 >> 2, tj1 = t1 & 3;
  v8f acc0 = {}, acc1 = {};

  for (int kc = 0; kc < 2048; kc += 64) {
    {
      const float* p = X + ((size_t)(n * 512 + ib * 64 + rr)) * HW + hw0 + kc + c4;
      __builtin_prefetch(p + 64, 0, 1);            // global_prefetch_b8 for next chunk
      f32x4 v0 = *(const f32x4*)p, v1 = *(const f32x4*)(p + 4);
      f32x4 v2 = *(const f32x4*)(p + 8), v3 = *(const f32x4*)(p + 12);
      bf16_t* dst = &As[rr][c4];
#pragma unroll
      for (int k = 0; k < 4; ++k) { dst[k] = f2bf(v0[k]); dst[4+k] = f2bf(v1[k]);
                                    dst[8+k] = f2bf(v2[k]); dst[12+k] = f2bf(v3[k]); }
    }
    {
      const float* p = X + ((size_t)(n * 512 + jb * 64 + rr)) * HW + hw0 + kc + c4;
      f32x4 v0 = *(const f32x4*)p, v1 = *(const f32x4*)(p + 4);
      f32x4 v2 = *(const f32x4*)(p + 8), v3 = *(const f32x4*)(p + 12);
      bf16_t* dst = &Bs[rr][c4];
#pragma unroll
      for (int k = 0; k < 4; ++k) { dst[k] = f2bf(v0[k]); dst[4+k] = f2bf(v1[k]);
                                    dst[8+k] = f2bf(v2[k]); dst[12+k] = f2bf(v3[k]); }
    }
    __syncthreads();
#pragma unroll
    for (int kk = 0; kk < 64; kk += 32) {
      v16bf a0 = frag_a_lds(&As[ti0 * 16][0], 64, kk, lane);
      v16bf b0 = frag_b_lds(&Bs[tj0 * 16][0], 64, kk, lane);
      acc0 = wmma_bf16(a0, b0, acc0);
      v16bf a1 = frag_a_lds(&As[ti1 * 16][0], 64, kk, lane);
      v16bf b1 = frag_b_lds(&Bs[tj1 * 16][0], 64, kk, lane);
      acc1 = wmma_bf16(a1, b1, acc1);
    }
    __syncthreads();
  }
  int half8 = ((lane >> 4) & 1) * 8;
  int j0 = jb * 64 + tj0 * 16 + (lane & 15);
  int i0 = ib * 64 + ti0 * 16 + half8;
#pragma unroll
  for (int r = 0; r < 8; ++r) atomicAdd(&G[(size_t)(i0 + r) * 512 + j0], acc0[r]);
  int j1 = jb * 64 + tj1 * 16 + (lane & 15);
  int i1 = ib * 64 + ti1 * 16 + half8;
#pragma unroll
  for (int r = 0; r < 8; ++r) atomicAdd(&G[(size_t)(i1 + r) * 512 + j1], acc1[r]);
}

// mu_d = w_d . mean(x)   (b1 cancels in (h - mu)),  rsig_d = rsqrt(var + eps)
__global__ void stats_kernel(const float* Gws, const float* S1ws,
                             const float* sW1, const float* tW1,
                             float* MU, float* RS) {
  int d = blockIdx.x, net = blockIdx.y;
  const float* G  = Gws + (size_t)net * 512 * 512;
  const float* S1 = S1ws + net * 512;
  const float* W1 = net ? tW1 : sW1;
  __shared__ float wbuf[512], mxbuf[512], red[256];
  int t = threadIdx.x;
  const float invP = 1.0f / 131072.0f;
  for (int c = t; c < 512; c += 256) { wbuf[c] = W1[(size_t)d * 512 + c]; mxbuf[c] = S1[c] * invP; }
  __syncthreads();
  float q = 0.f, tv = 0.f;
  for (int c = t; c < 512; c += 256) {
    float wc = wbuf[c];
    const float* gr = G + (size_t)c * 512;
    float inner = 0.f;
    for (int c2 = 0; c2 < 512; c2 += 4) {
      f32x4 g = *(const f32x4*)(gr + c2);
      inner += g[0]*wbuf[c2] + g[1]*wbuf[c2+1] + g[2]*wbuf[c2+2] + g[3]*wbuf[c2+3];
    }
    q  += wc * inner;
    tv += wc * mxbuf[c];
  }
  q  = block_reduce_sum(q, red);
  tv = block_reduce_sum(tv, red);
  if (t == 0) {
    float var = q * invP - tv * tv;
    MU[net * 512 + d] = tv;                               // no b1 (cancels)
    RS[net * 512 + d] = rsqrtf(fmaxf(var, 0.0f) + 1e-5f);
  }
}

// Per-sampled-pixel embedding: 16 pixels per block (anchor order r = v*152+a).
__global__ void embed_kernel(const float* pS, const float* pT,
                             const int* sbatch, const int* sidx,
                             const float* sW1, const float* sG, const float* sBe,
                             const float* sW2, const float* sB2,
                             const float* tW1, const float* tG, const float* tBe,
                             const float* tW2, const float* tB2,
                             const float* MU, const float* RS, bf16_t* Fout) {
  int net = blockIdx.y;
  const float* X   = net ? pT  : pS;
  const float* W1  = net ? tW1 : sW1;
  const float* gam = net ? tG  : sG;
  const float* bet = net ? tBe : sBe;
  const float* W2  = net ? tW2 : sW2;
  const float* B2  = net ? tB2 : sB2;
  const float* mu  = MU + net * 512;
  const float* rs  = RS + net * 512;
  bf16_t* Fo = Fout + (size_t)net * 3952 * 256;

  __shared__ bf16_t Xl[16][512];
  __shared__ bf16_t Hl[16][512];
  __shared__ float  Zl[16][256];
  __shared__ float  nrm2[16];
  __shared__ int    pn[16], phw[16];

  int t = threadIdx.x, lane = t & 31, wave = t >> 5;
  int b = blockIdx.x;
  if (t < 16) {
    int r = b * 16 + t;
    int v = r / 152, a = r - v * 152;
    pn[t]  = sbatch[a];
    phw[t] = sidx[a * 26 + v];
    nrm2[t] = 0.f;
  }
  __syncthreads();
  {
    int px = t >> 4, c0 = (t & 15) * 32;
    const float* bp = X + ((size_t)pn[px] * 512) * 16384 + phw[px];
    for (int k = 0; k < 32; ++k) Xl[px][c0 + k] = f2bf(bp[(size_t)(c0 + k) * 16384]);
  }
  __syncthreads();

  int mrow = lane & 15, half8 = ((lane >> 4) & 1) * 8;

  // GEMM1: H(512x16) = W1 * X, each wave owns 4 row-tiles.
  v8f acc[4] = {};
  for (int kc = 0; kc < 512; kc += 32) {
    v16bf bfrag = frag_b_lds(&Xl[0][0], 512, kc, lane);
#pragma unroll
    for (int q = 0; q < 4; ++q) {
      int d0 = (wave * 4 + q) * 16;
      v16bf afrag = frag_a_gf32(W1 + (size_t)(d0 + mrow) * 512, kc, lane);
      acc[q] = wmma_bf16(afrag, bfrag, acc[q]);
    }
  }
#pragma unroll
  for (int q = 0; q < 4; ++q) {
    int d0 = (wave * 4 + q) * 16 + half8;
    v8bf hv;
#pragma unroll
    for (int r = 0; r < 8; ++r) {
      int d = d0 + r;
      float h = (acc[q][r] - mu[d]) * rs[d] * gam[d] + bet[d];
      hv[r] = f2bf(fmaxf(h, 0.0f));
    }
    *(v8bf*)&Hl[mrow][d0] = hv;
  }
  __syncthreads();

  // GEMM2: Z(256x16) = W2 * relu(BN(H)), each wave owns 2 row-tiles.
  v8f acc2[2] = {};
  for (int kc = 0; kc < 512; kc += 32) {
    v16bf bfrag = frag_b_lds(&Hl[0][0], 512, kc, lane);
#pragma unroll
    for (int q = 0; q < 2; ++q) {
      int d0 = (wave * 2 + q) * 16;
      v16bf afrag = frag_a_gf32(W2 + (size_t)(d0 + mrow) * 512, kc, lane);
      acc2[q] = wmma_bf16(afrag, bfrag, acc2[q]);
    }
  }
#pragma unroll
  for (int q = 0; q < 2; ++q) {
    int d0 = (wave * 2 + q) * 16 + half8;
#pragma unroll
    for (int r = 0; r < 8; ++r) Zl[mrow][d0 + r] = acc2[q][r] + B2[d0 + r];
  }
  __syncthreads();
  {
    int px = t >> 4, j = t & 15;
    float s = 0.f;
#pragma unroll
    for (int k = 0; k < 16; ++k) { float z = Zl[px][j * 16 + k]; s += z * z; }
    atomicAdd(&nrm2[px], s);
  }
  __syncthreads();
  {
    int px = t >> 4, j = t & 15;
    float inv = 1.0f / fmaxf(sqrtf(nrm2[px]), 1e-12f);
    size_t base = (size_t)(b * 16 + px) * 256 + j * 16;
#pragma unroll
    for (int k = 0; k < 16; ++k) Fo[base + k] = f2bf(Zl[px][j * 16 + k] * inv);
  }
}

// L = FS FT^T / TEMP (3952x3952), also writes L^T for the reverse direction.
__global__ void logits_kernel(const bf16_t* F, float* L, float* LT) {
  const bf16_t* FS = F;
  const bf16_t* FT = F + (size_t)3952 * 256;
  __shared__ bf16_t Sa[64][256];
  __shared__ bf16_t Sb[64][256];
  int i0 = blockIdx.x * 64, j0 = blockIdx.y * 64;
  int t = threadIdx.x, lane = t & 31, wave = t >> 5;
  {
    int rr = t >> 2, c0 = (t & 3) * 64;
    uint4 z = {0, 0, 0, 0};
    uint4* dA = (uint4*)&Sa[rr][c0];
    uint4* dB = (uint4*)&Sb[rr][c0];
    const uint4* sA = (i0 + rr < 3952) ? (const uint4*)(FS + (size_t)(i0 + rr) * 256 + c0) : nullptr;
    const uint4* sB = (j0 + rr < 3952) ? (const uint4*)(FT + (size_t)(j0 + rr) * 256 + c0) : nullptr;
#pragma unroll
    for (int q = 0; q < 8; ++q) dA[q] = sA ? sA[q] : z;
#pragma unroll
    for (int q = 0; q < 8; ++q) dB[q] = sB ? sB[q] : z;
  }
  __syncthreads();
  int t0 = wave * 2, t1 = t0 + 1;
  int ti0 = t0 >> 2, tj0 = t0 & 3, ti1 = t1 >> 2, tj1 = t1 & 3;
  v8f a0 = {}, a1 = {};
  for (int kc = 0; kc < 256; kc += 32) {
    v16bf af0 = frag_a_lds(&Sa[ti0 * 16][0], 256, kc, lane);
    v16bf bf0 = frag_b_lds(&Sb[tj0 * 16][0], 256, kc, lane);
    a0 = wmma_bf16(af0, bf0, a0);
    v16bf af1 = frag_a_lds(&Sa[ti1 * 16][0], 256, kc, lane);
    v16bf bf1 = frag_b_lds(&Sb[tj1 * 16][0], 256, kc, lane);
    a1 = wmma_bf16(af1, bf1, a1);
  }
  const float sc = 1.0f / 0.07f;
  int half8 = ((lane >> 4) & 1) * 8, nn = lane & 15;
  {
    int ib = i0 + ti0 * 16 + half8, j = j0 + tj0 * 16 + nn;
    if (j < 3952)
#pragma unroll
      for (int r = 0; r < 8; ++r) {
        int i = ib + r;
        if (i < 3952) { float v = a0[r] * sc; L[(size_t)i * 3952 + j] = v; LT[(size_t)j * 3952 + i] = v; }
      }
  }
  {
    int ib = i0 + ti1 * 16 + half8, j = j0 + tj1 * 16 + nn;
    if (j < 3952)
#pragma unroll
      for (int r = 0; r < 8; ++r) {
        int i = ib + r;
        if (i < 3952) { float v = a1[r] * sc; L[(size_t)i * 3952 + j] = v; LT[(size_t)j * 3952 + i] = v; }
      }
  }
}

// Per-row supervised-contrastive reduction (one direction; matrix row-major).
__global__ void row_loss_kernel(const float* Lm, const int* labels, float* acc) {
  const int M = 3952;
  int i = blockIdx.x, t = threadIdx.x;
  __shared__ int lab[152];
  __shared__ float red[256];
  __shared__ float sh_max, sh_neg;
  for (int a = t; a < 152; a += 256) lab[a] = labels[a];
  __syncthreads();
  int li = lab[i % 152];
  const float* row = Lm + (size_t)i * M;

  float m = -3.4e38f;
  for (int j = t; j < M; j += 256) m = fmaxf(m, row[j]);
  red[t] = m; __syncthreads();
  for (int s = 128; s > 0; s >>= 1) { if (t < s) red[t] = fmaxf(red[t], red[t + s]); __syncthreads(); }
  if (t == 0) sh_max = red[0];
  __syncthreads();
  float mx = sh_max;

  float neg = 0.f;
  for (int j = t; j < M; j += 256) {
    int lj = lab[j % 152];
    if (lj != li) neg += __expf(row[j] - mx);
  }
  red[t] = neg; __syncthreads();
  for (int s = 128; s > 0; s >>= 1) { if (t < s) red[t] += red[t + s]; __syncthreads(); }
  if (t == 0) sh_neg = red[0];
  __syncthreads();
  float ng = sh_neg;

  float pos = 0.f, cnt = 0.f;
  for (int j = t; j < M; j += 256) {
    if (j == i) continue;
    int lj = lab[j % 152];
    if (lj == li) {
      float s = row[j] - mx;
      pos += s - __logf(__expf(s) + ng);
      cnt += 1.f;
    }
  }
  red[t] = pos; __syncthreads();
  for (int s = 128; s > 0; s >>= 1) { if (t < s) red[t] += red[t + s]; __syncthreads(); }
  float psum = red[0]; __syncthreads();
  red[t] = cnt; __syncthreads();
  for (int s = 128; s > 0; s >>= 1) { if (t < s) red[t] += red[t + s]; __syncthreads(); }
  if (t == 0) {
    float mlp = psum / (red[0] + 1e-6f);
    atomicAdd(acc, (-(0.07f / 0.07f) * mlp) / (float)M);
  }
}

__global__ void final_kernel(const float* acc, float* out) {
  if (threadIdx.x == 0 && blockIdx.x == 0) out[0] = 0.5f * (acc[0] + acc[1]);
}

// ---------------------------------------------------------------------------
extern "C" void kernel_launch(void* const* d_in, const int* in_sizes, int n_in,
                              void* d_out, int out_size, void* d_ws, size_t ws_size,
                              hipStream_t stream) {
  (void)in_sizes; (void)n_in; (void)out_size; (void)ws_size;
  const float* pS     = (const float*)d_in[0];
  const float* pT     = (const float*)d_in[1];
  const int*   sbatch = (const int*)d_in[2];
  const int*   sidx   = (const int*)d_in[3];
  const int*   labels = (const int*)d_in[4];
  const float* sW1 = (const float*)d_in[5];
  const float* sG  = (const float*)d_in[7];
  const float* sBe = (const float*)d_in[8];
  const float* sW2 = (const float*)d_in[9];
  const float* sB2 = (const float*)d_in[10];
  const float* tW1 = (const float*)d_in[11];
  const float* tG  = (const float*)d_in[13];
  const float* tBe = (const float*)d_in[14];
  const float* tW2 = (const float*)d_in[15];
  const float* tB2 = (const float*)d_in[16];
  // (b1 inputs d_in[6], d_in[12] cancel inside BN and are unused.)

  float* ws  = (float*)d_ws;
  float* G   = ws;                     // 2 * 512*512
  float* S1  = G  + 2 * 512 * 512;     // 2 * 512
  float* MU  = S1 + 1024;              // 2 * 512
  float* RS  = MU + 1024;              // 2 * 512
  float* ACC = RS + 1024;              // 8 (2 used)
  bf16_t* FB = (bf16_t*)(ACC + 8);     // 2 * 3952 * 256 bf16
  float* Lm  = (float*)((char*)FB + (size_t)2 * 3952 * 256 * sizeof(bf16_t));
  float* LTm = Lm + (size_t)3952 * 3952;

  size_t nz = 2 * 512 * 512 + 1024 + 1024 + 1024 + 8;
  zero_kernel<<<dim3(512), 256, 0, stream>>>(ws, nz);
  sum_kernel<<<dim3(4096, 2), 256, 0, stream>>>(pS, pT, S1);
  cov_kernel<<<dim3(64, 64, 2), 256, 0, stream>>>(pS, pT, G);
  stats_kernel<<<dim3(512, 2), 256, 0, stream>>>(G, S1, sW1, tW1, MU, RS);
  embed_kernel<<<dim3(247, 2), 256, 0, stream>>>(pS, pT, sbatch, sidx,
      sW1, sG, sBe, sW2, sB2, tW1, tG, tBe, tW2, tB2, MU, RS, FB);
  logits_kernel<<<dim3(62, 62), 256, 0, stream>>>(FB, Lm, LTm);
  row_loss_kernel<<<dim3(3952), 256, 0, stream>>>(Lm, labels, ACC + 0);
  row_loss_kernel<<<dim3(3952), 256, 0, stream>>>(LTm, labels, ACC + 1);
  final_kernel<<<1, 32, 0, stream>>>(ACC, (float*)d_out);
}
